// LocalGlobalBlock_8907762171978
// MI455X (gfx1250) — matrix operand order repaired and checked by reading, not compiled, not run on hardware
//
#include <hip/hip_runtime.h>

// ---------------------------------------------------------------------------
// MI455X (gfx1250) implementation of the LocalGlobal transformer block.
// All GEMMs run on v_wmma_f32_16x16x32_bf16 (bf16 inputs, f32 accumulate).
// GEMM main loops are software-pipelined AND 2x-unrolled so the two fragment
// buffers alternate roles positionally (no cross-iteration register copies),
// letting loads stay in flight across the WMMAs with partial s_wait_loadcnt.
// Attention: flash-style online softmax per (head, 16-row query tile) with
// WMMA for both QK^T and P*V; P is re-fragmented through LDS.
// ---------------------------------------------------------------------------

typedef __attribute__((ext_vector_type(16))) __bf16 v16bf;
typedef __attribute__((ext_vector_type(8)))  __bf16 v8bf;
typedef __attribute__((ext_vector_type(8)))  float  v8f;

#define H_TOK   3072
#define G_TOKN  16
#define SEQL    3088        // H_TOK + G_TOKN, exactly 193 tiles of 16
#define DMODEL  512
#define DFF     2048
#define QTILES  193
#define RADIUS_ 128

// ---------------------------- WMMA helpers --------------------------------

__device__ __forceinline__ v8f wmma_bf16f32(v16bf a, v16bf b, v8f c) {
  // D = A(16x32 bf16) * B(32x16 bf16) + C(16x16 f32)
  return __builtin_amdgcn_wmma_f32_16x16x32_bf16(
      /*neg_a=*/false, a, /*neg_b=*/false, b,
      /*c_mod=*/(short)0, c, /*reuse_a=*/false, /*reuse_b=*/false);
}

// A fragment: 16x32 tile starting at (m0, k0) of row-major A (lda = K).
// contiguous-k A order used here (same as B): lanes 0-15 row=lane, K = 0..15;
// lanes 16-31 row=lane-16, K = 16..31.
__device__ __forceinline__ v16bf load_a_frag(const __bf16* A, int lda, int m0, int k0) {
  const int lane = threadIdx.x & 31;
  const int half = lane >> 4;
  const __bf16* p = A + (size_t)(m0 + (lane & 15)) * lda + k0 + half * 16;
  union { v16bf v; v8bf h[2]; } u;
  u.h[0] = *(const v8bf*)(p);
  u.h[1] = *(const v8bf*)(p + 8);
  return u.v;
}

// B fragment: B_mat[k, n] = W[n, k] with W row-major N x K (ldw = K).
// Lanes 0-15: col=lane, K = k0..k0+15; lanes 16-31: col=lane-16, K=k0+16..31.
__device__ __forceinline__ v16bf load_b_frag(const __bf16* W, int ldw, int n0, int k0) {
  const int lane = threadIdx.x & 31;
  const int half = lane >> 4;
  return *(const v16bf*)(W + (size_t)(n0 + (lane & 15)) * ldw + k0 + half * 16);
}

__device__ __forceinline__ float gelu_exact(float x) {
  return 0.5f * x * (1.0f + erff(x * 0.70710678118654752f));
}

// ------------------------- small prep kernels ------------------------------

__global__ void build_seq_kernel(const float* __restrict__ x,
                                 const float* __restrict__ gtok,
                                 float* __restrict__ seqf,
                                 __bf16* __restrict__ seqb) {
  int i = blockIdx.x * blockDim.x + threadIdx.x;        // SEQL*DMODEL threads
  int row = i >> 9;
  int col = i & 511;
  float v = (row < H_TOK) ? x[i] : gtok[(row - H_TOK) * DMODEL + col];
  seqf[i] = v;
  seqb[i] = (__bf16)v;
}

__global__ void cvt_kernel(const float* __restrict__ in,
                           __bf16* __restrict__ out, int n) {
  int i = blockIdx.x * blockDim.x + threadIdx.x;
  if (i < n) out[i] = (__bf16)in[i];
}

// ------------------------------ GEMM ---------------------------------------
// C[M,N] = A[M,K] @ W[N,K]^T + bias ; one wave owns a 16x64 output tile.
// Double-buffered + 2x unrolled: no register copies, loads overlap WMMAs.
// Requires K % 64 == 0 (true for all GEMMs here: K = 512 or 2048).

__global__ __launch_bounds__(256) void gemm_bf16_kernel(
    const __bf16* __restrict__ A, const __bf16* __restrict__ W,
    const float* __restrict__ bias, __bf16* __restrict__ outb,
    float* __restrict__ outf, int M, int N, int K, int act) {
  const int wave = (blockIdx.x * blockDim.x + threadIdx.x) >> 5;
  const int ntn  = N >> 6;
  const int mt   = wave / ntn;
  const int nt   = wave - mt * ntn;
  if (mt >= (M >> 4)) return;                  // whole-wave uniform exit
  const int m0 = mt << 4, n0 = nt << 6;
  const int lane = threadIdx.x & 31;

  const int B = K >> 5;                        // # of 32-wide k blocks (even)
  v8f acc[4] = {};
  v16bf a0, a1, b0[4], b1[4];

  // prologue: block 0 -> buffer 0
  a0 = load_a_frag(A, K, m0, 0);
#pragma unroll
  for (int t = 0; t < 4; t++) b0[t] = load_b_frag(W, K, n0 + t * 16, 0);

  int i = 1;
  for (; i + 1 < B; i += 2) {
    const int ka = i << 5;
    const int kb = (i + 1) << 5;
    // load block i -> buffer 1, compute block i-1 from buffer 0
    a1 = load_a_frag(A, K, m0, ka);
#pragma unroll
    for (int t = 0; t < 4; t++) b1[t] = load_b_frag(W, K, n0 + t * 16, ka);
    __builtin_prefetch(A + (size_t)(m0 + (lane & 15)) * K + kb, 0, 1);
#pragma unroll
    for (int t = 0; t < 4; t++) acc[t] = wmma_bf16f32(a0, b0[t], acc[t]);
    // load block i+1 -> buffer 0, compute block i from buffer 1
    a0 = load_a_frag(A, K, m0, kb);
#pragma unroll
    for (int t = 0; t < 4; t++) b0[t] = load_b_frag(W, K, n0 + t * 16, kb);
#pragma unroll
    for (int t = 0; t < 4; t++) acc[t] = wmma_bf16f32(a1, b1[t], acc[t]);
  }
  // tail: exactly one block remains (B even -> i == B-1 here)
  {
    const int ka = i << 5;
    a1 = load_a_frag(A, K, m0, ka);
#pragma unroll
    for (int t = 0; t < 4; t++) b1[t] = load_b_frag(W, K, n0 + t * 16, ka);
#pragma unroll
    for (int t = 0; t < 4; t++) acc[t] = wmma_bf16f32(a0, b0[t], acc[t]);
#pragma unroll
    for (int t = 0; t < 4; t++) acc[t] = wmma_bf16f32(a1, b1[t], acc[t]);
  }

  const int cb = lane & 15;
  const int ro = (lane >> 4) << 3;             // +8 rows for upper half-wave
#pragma unroll
  for (int t = 0; t < 4; t++) {
    const int col = n0 + t * 16 + cb;
    const float bv = bias ? bias[col] : 0.0f;
#pragma unroll
    for (int r = 0; r < 8; r++) {
      const int row = m0 + ro + r;
      float v = acc[t][r] + bv;
      if (act) v = gelu_exact(v);
      if (outf) outf[(size_t)row * N + col] = v;
      if (outb) outb[(size_t)row * N + col] = (__bf16)v;
    }
  }
}

// --------------------------- attention -------------------------------------
// One block per 16-row query tile; 8 waves = 8 heads. Flash-style streaming
// over 32-key blocks (two 16-key tiles, second optionally masked off).

__global__ __launch_bounds__(256) void attn_kernel(const __bf16* __restrict__ qkv,
                                                   __bf16* __restrict__ attn) {
  __shared__ __bf16 pbuf_all[8 * 16 * 32];     // per-wave P staging (1 KB each)
  const int head = threadIdx.x >> 5;
  const int lane = threadIdx.x & 31;
  const int qt   = blockIdx.x;
  const int m0   = qt << 4;
  const int cb   = lane & 15;
  const int half = lane >> 4;
  const int ro   = half << 3;
  __bf16* pb = pbuf_all + head * 512;

  const int LDQ  = 3 * DMODEL;                 // qkv row stride (1536)
  const int qcol = head * 64;
  const int kcol = DMODEL + head * 64;
  const int vcol = 2 * DMODEL + head * 64;

  const v16bf aq0 = load_a_frag(qkv, LDQ, m0, qcol);
  const v16bf aq1 = load_a_frag(qkv, LDQ, m0, qcol + 32);

  float mrow[8], lrow[8];
  v8f acc[4] = {};
#pragma unroll
  for (int r = 0; r < 8; r++) { mrow[r] = -__builtin_inff(); lrow[r] = 0.0f; }

  auto process = [&](int j0, bool has2) {
    // hoist all K-fragment loads so they overlap the WMMA chain
    const v16bf bk00 = load_b_frag(qkv, LDQ, j0, kcol);
    const v16bf bk01 = load_b_frag(qkv, LDQ, j0, kcol + 32);
    v16bf bk10, bk11;
    if (has2) {
      bk10 = load_b_frag(qkv, LDQ, j0 + 16, kcol);
      bk11 = load_b_frag(qkv, LDQ, j0 + 16, kcol + 32);
    }
    // S = Q(16x64) @ K^T(64x32) * 1/sqrt(64), as two 16x16 tiles
    v8f c0 = {}, c1 = {};
    c0 = wmma_bf16f32(aq0, bk00, c0);
    c0 = wmma_bf16f32(aq1, bk01, c0);
    if (has2) {
      c1 = wmma_bf16f32(aq0, bk10, c1);
      c1 = wmma_bf16f32(aq1, bk11, c1);
    }
    const int kj0 = j0 + cb;
    const int kj1 = kj0 + 16;
#pragma unroll
    for (int r = 0; r < 8; r++) {
      const int qi = m0 + ro + r;
      float s0 = c0[r] * 0.125f;
      const bool al0 = ((qi - kj0 <= RADIUS_) && (kj0 - qi <= RADIUS_)) ||
                       (qi >= H_TOK) || (kj0 >= H_TOK);
      s0 = al0 ? s0 : -__builtin_inff();
      float s1 = -__builtin_inff();
      if (has2) {
        const bool al1 = ((qi - kj1 <= RADIUS_) && (kj1 - qi <= RADIUS_)) ||
                         (qi >= H_TOK) || (kj1 >= H_TOK);
        const float t = c1[r] * 0.125f;
        s1 = al1 ? t : -__builtin_inff();
      }
      // row max over the 16 lanes holding this row (width-16 xor shuffles)
      float mx = fmaxf(s0, s1);
#pragma unroll
      for (int off = 1; off < 16; off <<= 1) mx = fmaxf(mx, __shfl_xor(mx, off, 16));
      const float mo = mrow[r];
      const float mn = fmaxf(mo, mx);
      float alpha, p0, p1;
      if (mn == -__builtin_inff()) {           // fully-masked block for this row
        alpha = 1.0f; p0 = 0.0f; p1 = 0.0f;
      } else {
        alpha = __expf(mo - mn);               // exp(-inf)=0 on first hit
        p0 = __expf(s0 - mn);                  // exp(-inf)=0 for masked cols
        p1 = __expf(s1 - mn);
      }
      float rs = p0 + p1;
#pragma unroll
      for (int off = 1; off < 16; off <<= 1) rs += __shfl_xor(rs, off, 16);
      lrow[r] = lrow[r] * alpha + rs;
      mrow[r] = mn;
#pragma unroll
      for (int t = 0; t < 4; t++) acc[t][r] *= alpha;
      // stage P row-major 16x32 bf16 for A-fragment reload
      pb[(ro + r) * 32 + cb]      = (__bf16)p0;
      pb[(ro + r) * 32 + 16 + cb] = (__bf16)p1;
    }
    // reload P in the 16-bit A-matrix layout (LDS ops are in-order per wave)
    union { v16bf v; v8bf h[2]; } up;
    const __bf16* pr = pb + cb * 32 + half * 16;
    up.h[0] = *(const v8bf*)(pr);
    up.h[1] = *(const v8bf*)(pr + 8);
    const v16bf pa = up.v;
    // O += P(16x32) @ V(32x64)
#pragma unroll
    for (int t = 0; t < 4; t++) {
      v16bf bv;
      const int dim = vcol + t * 16 + cb;
#pragma unroll
      for (int kk = 0; kk < 16; kk++) {
        int row = j0 + half * 16 + kk;
        row = row > (SEQL - 1) ? (SEQL - 1) : row;   // padded keys have P==0
        bv[kk] = qkv[(size_t)row * LDQ + dim];
      }
      acc[t] = wmma_bf16f32(pa, bv, acc[t]);
    }
  };

  int jt0, jt1;
  if (qt == QTILES - 1) { jt0 = 0; jt1 = QTILES - 1; }   // global queries: all keys
  else {
    jt0 = qt - 8; if (jt0 < 0) jt0 = 0;                  // band covers qi±128
    jt1 = qt + 8; if (jt1 > QTILES - 1) jt1 = QTILES - 1;
  }
  for (int jb = jt0; jb <= jt1; jb += 2) process(jb << 4, (jb + 1) <= jt1);
  if (qt != QTILES - 1 && jt1 < QTILES - 1) process((QTILES - 1) << 4, false);

#pragma unroll
  for (int r = 0; r < 8; r++) {
    const float inv = 1.0f / lrow[r];
    const int row = m0 + ro + r;
#pragma unroll
    for (int t = 0; t < 4; t++)
      attn[(size_t)row * DMODEL + head * 64 + t * 16 + cb] =
          (__bf16)(acc[t][r] * inv);
  }
}

// ----------------------- residual + LayerNorm ------------------------------
// One block (256 threads = 8 waves) per row of 512.

__global__ __launch_bounds__(256) void add_ln_kernel(
    const float* __restrict__ res, const float* __restrict__ y,
    const float* __restrict__ g, const float* __restrict__ b,
    float* __restrict__ outf, __bf16* __restrict__ outb) {
  __shared__ float red[8];
  const int row = blockIdx.x;
  const int tid = threadIdx.x;
  const size_t base = (size_t)row * DMODEL;
  float x0 = res[base + tid] + y[base + tid];
  float x1 = res[base + 256 + tid] + y[base + 256 + tid];

  float s = x0 + x1;
#pragma unroll
  for (int off = 16; off > 0; off >>= 1) s += __shfl_xor(s, off, 32);
  if ((tid & 31) == 0) red[tid >> 5] = s;
  __syncthreads();
  const float mean = (red[0] + red[1] + red[2] + red[3] +
                      red[4] + red[5] + red[6] + red[7]) * (1.0f / 512.0f);
  __syncthreads();

  const float d0 = x0 - mean, d1 = x1 - mean;
  s = d0 * d0 + d1 * d1;
#pragma unroll
  for (int off = 16; off > 0; off >>= 1) s += __shfl_xor(s, off, 32);
  if ((tid & 31) == 0) red[tid >> 5] = s;
  __syncthreads();
  const float var = (red[0] + red[1] + red[2] + red[3] +
                     red[4] + red[5] + red[6] + red[7]) * (1.0f / 512.0f);
  const float rstd = rsqrtf(var + 1e-5f);

  const float o0 = d0 * rstd * g[tid] + b[tid];
  const float o1 = d1 * rstd * g[256 + tid] + b[256 + tid];
  if (outf) { outf[base + tid] = o0; outf[base + 256 + tid] = o1; }
  if (outb) { outb[base + tid] = (__bf16)o0; outb[base + 256 + tid] = (__bf16)o1; }
}

// ------------------------------ driver -------------------------------------

extern "C" void kernel_launch(void* const* d_in, const int* in_sizes, int n_in,
                              void* d_out, int out_size, void* d_ws, size_t ws_size,
                              hipStream_t stream) {
  const float* x    = (const float*)d_in[0];
  // d_in[1] = h (always 3072 here; dimensions are compile-time constants)
  const float* Wqkv = (const float*)d_in[2];
  const float* bqkv = (const float*)d_in[3];
  const float* Wo   = (const float*)d_in[4];
  const float* bo   = (const float*)d_in[5];
  const float* gtok = (const float*)d_in[6];
  const float* ln1g = (const float*)d_in[7];
  const float* ln1b = (const float*)d_in[8];
  const float* W1   = (const float*)d_in[9];
  const float* b1   = (const float*)d_in[10];
  const float* W2   = (const float*)d_in[11];
  const float* b2   = (const float*)d_in[12];
  const float* ln2g = (const float*)d_in[13];
  const float* ln2b = (const float*)d_in[14];
  float* out = (float*)d_out;

  char* w = (char*)d_ws;
  auto alloc = [&](size_t bytes) {
    char* p = w;
    w += (bytes + 255) & ~(size_t)255;
    return p;
  };
  float*  seqf   = (float*) alloc((size_t)SEQL * DMODEL * 4);
  __bf16* seqb   = (__bf16*)alloc((size_t)SEQL * DMODEL * 2);
  __bf16* wqkvb  = (__bf16*)alloc((size_t)3 * DMODEL * DMODEL * 2);
  __bf16* wob    = (__bf16*)alloc((size_t)DMODEL * DMODEL * 2);
  __bf16* w1b    = (__bf16*)alloc((size_t)DFF * DMODEL * 2);
  __bf16* w2b    = (__bf16*)alloc((size_t)DMODEL * DFF * 2);
  __bf16* qkvb   = (__bf16*)alloc((size_t)SEQL * 3 * DMODEL * 2);
  __bf16* attnb  = (__bf16*)alloc((size_t)SEQL * DMODEL * 2);
  float*  attnof = (float*) alloc((size_t)SEQL * DMODEL * 4);
  float*  seq1f  = (float*) alloc((size_t)SEQL * DMODEL * 4);
  __bf16* seq1b  = (__bf16*)alloc((size_t)SEQL * DMODEL * 2);
  __bf16* ffn1b  = (__bf16*)alloc((size_t)SEQL * DFF * 2);
  float*  ffn2f  = (float*) alloc((size_t)SEQL * DMODEL * 4);

  // 1) build seq (concat tokens + globals), downconvert once to bf16
  build_seq_kernel<<<(SEQL * DMODEL) / 256, 256, 0, stream>>>(x, gtok, seqf, seqb);
  cvt_kernel<<<(3 * DMODEL * DMODEL + 255) / 256, 256, 0, stream>>>(Wqkv, wqkvb, 3 * DMODEL * DMODEL);
  cvt_kernel<<<(DMODEL * DMODEL + 255) / 256, 256, 0, stream>>>(Wo, wob, DMODEL * DMODEL);
  cvt_kernel<<<(DFF * DMODEL + 255) / 256, 256, 0, stream>>>(W1, w1b, DFF * DMODEL);
  cvt_kernel<<<(DMODEL * DFF + 255) / 256, 256, 0, stream>>>(W2, w2b, DMODEL * DFF);

  // 2) QKV projection: (3088x512) @ (1536x512)^T
  {
    int waves = (SEQL / 16) * ((3 * DMODEL) / 64);
    gemm_bf16_kernel<<<(waves + 7) / 8, 256, 0, stream>>>(
        seqb, wqkvb, bqkv, qkvb, (float*)nullptr, SEQL, 3 * DMODEL, DMODEL, 0);
  }

  // 3) local+global attention (WMMA flash)
  attn_kernel<<<QTILES, 256, 0, stream>>>(qkvb, attnb);

  // 4) Wo projection -> f32
  {
    int waves = (SEQL / 16) * (DMODEL / 64);
    gemm_bf16_kernel<<<(waves + 7) / 8, 256, 0, stream>>>(
        attnb, wob, bo, (__bf16*)nullptr, attnof, SEQL, DMODEL, DMODEL, 0);
  }

  // 5) residual + LN1 -> seq1 (f32 + bf16)
  add_ln_kernel<<<SEQL, 256, 0, stream>>>(seqf, attnof, ln1g, ln1b, seq1f, seq1b);

  // 6) FFN up + exact GELU -> bf16
  {
    int waves = (SEQL / 16) * (DFF / 64);
    gemm_bf16_kernel<<<(waves + 7) / 8, 256, 0, stream>>>(
        seq1b, w1b, b1, ffn1b, (float*)nullptr, SEQL, DFF, DMODEL, 1);
  }

  // 7) FFN down -> f32
  {
    int waves = (SEQL / 16) * (DMODEL / 64);
    gemm_bf16_kernel<<<(waves + 7) / 8, 256, 0, stream>>>(
        ffn1b, w2b, b2, (__bf16*)nullptr, ffn2f, SEQL, DMODEL, DFF, 0);
  }

  // 8) residual + LN2 straight into d_out (token rows then global rows)
  add_ln_kernel<<<SEQL, 256, 0, stream>>>(seq1f, ffn2f, ln2g, ln2b, out, (__bf16*)nullptr);
}
